// AutoformerEncoderLayer_82875688944219
// MI455X (gfx1250) — matrix operand order repaired and checked by reading, not compiled, not run on hardware
//
#include <hip/hip_runtime.h>
#include <math.h>

// ---------------------------------------------------------------------------
// Autoformer encoder layer for MI455X (gfx1250), wave32 + WMMA + TDM.
//   B=8, L=1024, D=512, H=8, d=64, k_top=20, decomp kernel=25
// Pipeline:
//   Q = x@Wq+bq ; K = x@Wk+bk ; V = x@Wv+bv          (WMMA f16 GEMMs, TDM A-tiles)
//   attn = AutoCorrelation(Q,K,V)                     (fused VALU kernel)
//   x_s  = decomp(attn + x)
//   h1   = relu(x_s@W1+b1) ; y = h1@W2+b2             (WMMA f16 GEMMs)
//   out  = decomp(y + x_s)
// ---------------------------------------------------------------------------

typedef _Float16 f16;
typedef __attribute__((ext_vector_type(16))) _Float16 v16h;
typedef __attribute__((ext_vector_type(8)))  _Float16 v8h;
typedef __attribute__((ext_vector_type(8)))  float    v8f;
typedef __attribute__((ext_vector_type(4)))  unsigned int u32x4;
typedef __attribute__((ext_vector_type(8)))  int          i32x8;
typedef __attribute__((ext_vector_type(4)))  int          i32x4;

#define Bsz 8
#define Lsz 1024
#define Dsz 512
#define NELEM (Bsz * Lsz * Dsz)   // 4194304
#define KTOP 20

// ===========================================================================
// TDM: issue a 2D tensor tile load (Global -> LDS), data_size = 4B.
// D# packing per CDNA5 ISA 8.3/8.4. LDS rows get pad: every 32 dwords,
// +8 dwords -> effective pitch 40 dwords (160B), killing pow2 bank conflicts.
// ===========================================================================
#define TDM_FLAGS_PAD40 ((2u << 16) | (1u << 20) | (4u << 22) | (7u << 25))

__device__ __forceinline__ void tdm_load_2d(unsigned lds_off, const void* gptr,
                                            unsigned tensor_d0, unsigned tensor_d1,
                                            unsigned tile_d0, unsigned tile_d1,
                                            unsigned stride0, unsigned flags1)
{
    unsigned long long ga = (unsigned long long)(size_t)gptr;
    u32x4 g0;
    g0[0] = 1u;                                   // count=1, user descriptor
    g0[1] = lds_off;                              // LDS byte address
    g0[2] = (unsigned)(ga & 0xFFFFFFFFu);         // global addr [31:0]
    g0[3] = (unsigned)((ga >> 32) & 0x01FFFFFFu)  // global addr [56:32]
          | (2u << 30);                           // type = 2 (image)
    i32x8 g1;
    g1[0] = (int)flags1;                          // wg_mask=0|data_size|pad cfg
    g1[1] = (int)((tensor_d0 & 0xFFFFu) << 16);   // barrier addr=0 | dim0 lo16
    g1[2] = (int)(((tensor_d0 >> 16) & 0xFFFFu) | ((tensor_d1 & 0xFFFFu) << 16));
    g1[3] = (int)(((tensor_d1 >> 16) & 0xFFFFu) | ((tile_d0 & 0xFFFFu) << 16));
    g1[4] = (int)(tile_d1 & 0xFFFFu);             // tile_dim1 | tile_dim2=0
    g1[5] = (int)stride0;                         // dim0 stride [31:0]
    g1[6] = 0;                                    // stride[47:32]=0, dim1_stride lo=0
    g1[7] = 0;
    i32x4 z4 = (i32x4){0, 0, 0, 0};               // groups 2/3 unused (2D tile)
    i32x8 z8 = (i32x8){0, 0, 0, 0, 0, 0, 0, 0};   // extra group (unused)
    __builtin_amdgcn_tensor_load_to_lds(g0, g1, z4, z4, z8, 0);
}

// ===========================================================================
// GEMM: C[M,512] = act(A[M,512] @ W[512,512] + bias), f16 WMMA, f32 accum.
// Block = 256 threads = 8 waves; block tile 128x128; wave tile 64x32.
// A tile (128x32 f32) staged by the Tensor Data Mover with padded rows
// (pitch 40 dwords); B tile staged transposed as f16 by the threads.
// ===========================================================================
__global__ __launch_bounds__(256)
void gemm_wmma_kernel(const float* __restrict__ A, const float* __restrict__ W,
                      const float* __restrict__ bias, float* __restrict__ C,
                      int relu)
{
    __shared__ __align__(16) float    sAf[128 * 40];  // TDM dest, pitch 40 f32
    __shared__ __align__(16) _Float16 sB[128][40];    // B transposed: [n][k]

    const int t    = threadIdx.x;
    const int lane = t & 31;
    const int wave = t >> 5;
    const int wm   = wave & 1;         // 2 waves along M
    const int wn   = wave >> 1;        // 4 waves along N
    const int m0   = blockIdx.y * 128;
    const int n0   = blockIdx.x * 128;
    const int lm   = lane & 15;
    const int lg   = lane >> 4;

    const unsigned ldsA = (unsigned)(unsigned long long)(size_t)&sAf[0];

    v8f acc[4][2];
#pragma unroll
    for (int i = 0; i < 4; ++i)
#pragma unroll
        for (int j = 0; j < 2; ++j)
            acc[i][j] = (v8f){0.f, 0.f, 0.f, 0.f, 0.f, 0.f, 0.f, 0.f};

    for (int kk = 0; kk < 512; kk += 32) {
        // ---- A tile via TDM: 128 rows x 32 f32, row stride 512 ----------
        if (wave == 0) {
            tdm_load_2d(ldsA, A + (size_t)m0 * 512 + kk,
                        /*tensor_d0=*/512, /*tensor_d1=*/8192,
                        /*tile_d0=*/32, /*tile_d1=*/128,
                        /*stride0=*/512, TDM_FLAGS_PAD40);
        }
        // ---- stage B (transposed): sB[n][k] = W[kk+k][n0+n] -------------
#pragma unroll
        for (int i = 0; i < 4; ++i) {
            int linear = t + i * 256;
            int kr = linear >> 5;              // 0..31
            int n4 = linear & 31;              // 0..31
            float4 v = *(const float4*)(W + (size_t)(kk + kr) * 512 + n0 + n4 * 4);
            sB[n4 * 4 + 0][kr] = (f16)v.x;
            sB[n4 * 4 + 1][kr] = (f16)v.y;
            sB[n4 * 4 + 2][kr] = (f16)v.z;
            sB[n4 * 4 + 3][kr] = (f16)v.w;
        }
        if (kk + 32 < 512)
            __builtin_prefetch(W + (size_t)(kk + 32 + (t & 31)) * 512 + n0, 0, 1);

        __builtin_amdgcn_s_wait_tensorcnt(0);  // wave0's TDM done (no-op others)
        __syncthreads();

        // ---- A fragments: 16x32 f16 from f32 LDS ------------------------
        // lane (m=lm, g=lg) holds A[m][g*8 + 0..7] and A[m][16+g*8 + 0..7]
        v16h af[4];
#pragma unroll
        for (int fa = 0; fa < 4; ++fa) {
            const float* pr = &sAf[(wm * 64 + fa * 16 + lm) * 40 + lg * 8];
            float4 x0 = *(const float4*)(pr + 0);
            float4 x1 = *(const float4*)(pr + 4);
            float4 x2 = *(const float4*)(pr + 16);
            float4 x3 = *(const float4*)(pr + 20);
            af[fa] = (v16h){(f16)x0.x, (f16)x0.y, (f16)x0.z, (f16)x0.w,
                            (f16)x1.x, (f16)x1.y, (f16)x1.z, (f16)x1.w,
                            (f16)x2.x, (f16)x2.y, (f16)x2.z, (f16)x2.w,
                            (f16)x3.x, (f16)x3.y, (f16)x3.z, (f16)x3.w};
        }
        // ---- B fragments: 32x16 f16, lane holds sB[n][g*8 + 0..7], sB[n][16+g*8 + 0..7] -----
        v16h bf[2];
#pragma unroll
        for (int fb = 0; fb < 2; ++fb) {
            const _Float16* pr = &sB[wn * 32 + fb * 16 + lm][lg * 8];
            v8h lo = *(const v8h*)pr;
            v8h hi = *(const v8h*)(pr + 16);
            bf[fb] = __builtin_shufflevector(lo, hi, 0, 1, 2, 3, 4, 5, 6, 7,
                                             8, 9, 10, 11, 12, 13, 14, 15);
        }
#pragma unroll
        for (int fa = 0; fa < 4; ++fa)
#pragma unroll
            for (int fb = 0; fb < 2; ++fb)
                acc[fa][fb] = __builtin_amdgcn_wmma_f32_16x16x32_f16(
                    false, af[fa], false, bf[fb], (short)0, acc[fa][fb],
                    false, false);
        __syncthreads();
    }

    // ---- epilogue: C/D layout: element r of lane -> C[g*8+r][n] ---------
#pragma unroll
    for (int fa = 0; fa < 4; ++fa) {
#pragma unroll
        for (int fb = 0; fb < 2; ++fb) {
            int n = n0 + wn * 32 + fb * 16 + lm;
            float bv = bias[n];
#pragma unroll
            for (int r = 0; r < 8; ++r) {
                int m = m0 + wm * 64 + fa * 16 + lg * 8 + r;
                float v = acc[fa][fb][r] + bv;
                if (relu) v = fmaxf(v, 0.f);
                C[(size_t)m * 512 + n] = v;
            }
        }
    }
}

// ===========================================================================
// AutoCorrelation: one block per (b, channel) column. Time-domain circular
// correlation corr[tau] = sum_t q[t] * k[(t - tau) mod L], then top-20 +
// softmax + saturating-shift aggregation of V. Everything stays in LDS.
// attn may alias Q (Q column fully consumed before the final store).
// ===========================================================================
__global__ __launch_bounds__(256)
void autocorr_kernel(const float* __restrict__ Q, const float* __restrict__ K,
                     const float* __restrict__ V, float* __restrict__ attn)
{
    const int ch = blockIdx.x;        // 0..4095
    const int b  = ch >> 9;
    const int c  = ch & 511;
    const int t  = threadIdx.x;

    __shared__ __align__(16) float kdup[2048];
    __shared__ __align__(16) float qv[1024];   // Q column, later V column
    __shared__ float corr[1024];
    __shared__ float redv[256];
    __shared__ int   redi[256];
    __shared__ float topv[KTOP];
    __shared__ int   topi[KTOP];
    __shared__ float wts[KTOP];

    const size_t colbase = (size_t)b * Lsz * Dsz + c;
    const float* Qc = Q + colbase;
    const float* Kc = K + colbase;
    const float* Vc = V + colbase;

    for (int i = t; i < 1024; i += 256) {
        float kv = Kc[(size_t)i * 512];
        qv[i] = Qc[(size_t)i * 512];
        kdup[i] = kv;
        kdup[i + 1024] = kv;
    }
    __syncthreads();

    // ---- circular correlation, 4 lags/thread, 4x unrolled, b128 LDS -----
    const int tau = t * 4;
    float a0 = 0.f, a1 = 0.f, a2 = 0.f, a3 = 0.f;
    const int base = 1024 - tau;               // multiple of 4; base-3 >= 1
    float k0 = kdup[base - 1];
    float k1 = kdup[base - 2];
    float k2 = kdup[base - 3];
    for (int tt = 0; tt < 1024; tt += 4) {
        float4 qq = *(const float4*)&qv[tt];
        float4 kw = *(const float4*)&kdup[base + tt];
        a0 = fmaf(qq.x, kw.x, a0); a1 = fmaf(qq.x, k0, a1);
        a2 = fmaf(qq.x, k1, a2);   a3 = fmaf(qq.x, k2, a3);
        a0 = fmaf(qq.y, kw.y, a0); a1 = fmaf(qq.y, kw.x, a1);
        a2 = fmaf(qq.y, k0, a2);   a3 = fmaf(qq.y, k1, a3);
        a0 = fmaf(qq.z, kw.z, a0); a1 = fmaf(qq.z, kw.y, a1);
        a2 = fmaf(qq.z, kw.x, a2); a3 = fmaf(qq.z, k0, a3);
        a0 = fmaf(qq.w, kw.w, a0); a1 = fmaf(qq.w, kw.z, a1);
        a2 = fmaf(qq.w, kw.y, a2); a3 = fmaf(qq.w, kw.x, a3);
        k0 = kw.w; k1 = kw.z; k2 = kw.y;
    }
    corr[tau + 0] = a0;
    corr[tau + 1] = a1;
    corr[tau + 2] = a2;
    corr[tau + 3] = a3;
    __syncthreads();

    // ---- iterative top-20 (stable: lower index wins ties) ---------------
    for (int p = 0; p < KTOP; ++p) {
        float bm = corr[tau];
        int   bi = tau;
#pragma unroll
        for (int j = 1; j < 4; ++j) {
            float cv = corr[tau + j];
            if (cv > bm) { bm = cv; bi = tau + j; }
        }
        redv[t] = bm;
        redi[t] = bi;
        __syncthreads();
        for (int s = 128; s > 0; s >>= 1) {
            if (t < s) {
                float ov = redv[t + s];
                int   oi = redi[t + s];
                if (ov > redv[t] || (ov == redv[t] && oi < redi[t])) {
                    redv[t] = ov;
                    redi[t] = oi;
                }
            }
            __syncthreads();
        }
        if (t == 0) {
            topv[p] = redv[0];
            topi[p] = redi[0];
            corr[redi[0]] = -3.0e38f;   // mask winner
        }
        __syncthreads();
    }

    // ---- softmax over the 20 selected weights ---------------------------
    if (t == 0) {
        float mx = topv[0];             // descending order -> max first
        float s = 0.f;
        for (int p = 0; p < KTOP; ++p) {
            float e = __expf(topv[p] - mx);
            wts[p] = e;
            s += e;
        }
        float inv = 1.f / s;
        for (int p = 0; p < KTOP; ++p) wts[p] *= inv;
    }
    __syncthreads();

    // ---- load V column (reuse qv) and aggregate -------------------------
    for (int i = t; i < 1024; i += 256) qv[i] = Vc[(size_t)i * 512];
    __syncthreads();

#pragma unroll
    for (int j = 0; j < 4; ++j) {
        int l = tau + j;
        float r = 0.f;
        for (int p = 0; p < KTOP; ++p) {
            int pos = topi[p] + l;
            if (pos > 1023) pos = 1023;   // saturating shift (clamp max=L-1)
            r = fmaf(wts[p], qv[pos], r);
        }
        attn[((size_t)b * Lsz + l) * Dsz + c] = r;
    }
}

// ===========================================================================
// Elementwise add: o = a + b
// ===========================================================================
__global__ __launch_bounds__(256)
void add_kernel(const float* __restrict__ a, const float* __restrict__ b,
                float* __restrict__ o)
{
    int i = blockIdx.x * 256 + threadIdx.x;
    if (i < NELEM) o[i] = a[i] + b[i];
}

// ===========================================================================
// Series decomposition residual: o = s - movavg25(s) with zero padding,
// count_include_pad semantics (divide by 25 always). Coalesced along D.
// ===========================================================================
__global__ __launch_bounds__(256)
void decomp_kernel(const float* __restrict__ s, float* __restrict__ o)
{
    int i = blockIdx.x * 256 + threadIdx.x;
    if (i >= NELEM) return;
    int d = i & 511;
    int l = (i >> 9) & 1023;
    int b = i >> 19;
    const float* col = s + (size_t)b * Lsz * Dsz + d;
    float sum = 0.f;
#pragma unroll
    for (int j = -12; j <= 12; ++j) {
        int ll = l + j;
        if (ll >= 0 && ll < 1024) sum += col[(size_t)ll * 512];
    }
    o[i] = s[i] - sum * (1.0f / 25.0f);
}

// ===========================================================================
// Host-side orchestration
// ===========================================================================
extern "C" void kernel_launch(void* const* d_in, const int* in_sizes, int n_in,
                              void* d_out, int out_size, void* d_ws, size_t ws_size,
                              hipStream_t stream)
{
    const float* x  = (const float*)d_in[0];
    const float* Wq = (const float*)d_in[1];
    const float* bq = (const float*)d_in[2];
    const float* Wk = (const float*)d_in[3];
    const float* bk = (const float*)d_in[4];
    const float* Wv = (const float*)d_in[5];
    const float* bv = (const float*)d_in[6];
    const float* W1 = (const float*)d_in[7];
    const float* b1 = (const float*)d_in[8];
    const float* W2 = (const float*)d_in[9];
    const float* b2 = (const float*)d_in[10];
    float* out = (float*)d_out;

    // Workspace layout: 4 buffers of B*L*D floats (64 MB total)
    float* bufQ = (float*)d_ws;            // Q  -> attn (in-place) -> s2
    float* bufK = bufQ + NELEM;            // K  -> y
    float* bufV = bufK + NELEM;            // V  -> s1 -> h1
    float* bufX = bufV + NELEM;            // x_s

    dim3 gGemm(4, 64);                     // N/128 x M/128  (M=8192, N=512)
    dim3 bGemm(256);
    dim3 gCorr(Bsz * Dsz);                 // 4096 channels
    dim3 gElem((NELEM + 255) / 256);

    // QKV projections
    gemm_wmma_kernel<<<gGemm, bGemm, 0, stream>>>(x, Wq, bq, bufQ, 0);
    gemm_wmma_kernel<<<gGemm, bGemm, 0, stream>>>(x, Wk, bk, bufK, 0);
    gemm_wmma_kernel<<<gGemm, bGemm, 0, stream>>>(x, Wv, bv, bufV, 0);

    // AutoCorrelation attention (attn overwrites Q in place, per-column safe)
    autocorr_kernel<<<gCorr, 256, 0, stream>>>(bufQ, bufK, bufV, bufQ);

    // x_s = decomp(attn + x)
    add_kernel<<<gElem, 256, 0, stream>>>(bufQ, x, bufV);
    decomp_kernel<<<gElem, 256, 0, stream>>>(bufV, bufX);

    // FFN: h1 = relu(x_s@W1+b1); y = h1@W2+b2
    gemm_wmma_kernel<<<gGemm, bGemm, 0, stream>>>(bufX, W1, b1, bufV, 1);
    gemm_wmma_kernel<<<gGemm, bGemm, 0, stream>>>(bufV, W2, b2, bufK, 0);

    // out = decomp(y + x_s)
    add_kernel<<<gElem, 256, 0, stream>>>(bufK, bufX, bufQ);
    decomp_kernel<<<gElem, 256, 0, stream>>>(bufQ, out);
}
